// MultiheadedAttention_6330781794994
// MI455X (gfx1250) — matrix operand order repaired and checked by reading, not compiled, not run on hardware
//
#include <hip/hip_runtime.h>

// ---------------------------------------------------------------- types
typedef __attribute__((ext_vector_type(16))) _Float16 v16h;
typedef __attribute__((ext_vector_type(8)))  float    v8f;
typedef __attribute__((ext_vector_type(4)))  int      v4i;

union AF { v16h h; v4i q[2]; };

#define Bn   2
#define Hn   16
#define Sn   2048
#define Dn   1024
#define DH   64

__device__ __forceinline__ v8f wmma16(const AF& a, const AF& b, v8f c) {
  return __builtin_amdgcn_wmma_f32_16x16x32_f16(
      false, a.h, false, b.h, (short)0, c, false, false);
}
__device__ __forceinline__ int pkh(float a, float b) {
  auto t = __builtin_amdgcn_cvt_pkrtz(a, b);  // __fp16 ext_vector(2), 4 bytes
  return __builtin_bit_cast(int, t);
}
// lane <-> lane^16 exchange (wave32), SWAPX16 via ds_swizzle
__device__ __forceinline__ int   swap16i(int v)   { return __builtin_amdgcn_ds_swizzle(v, 0x401F); }
__device__ __forceinline__ float swap16f(float v) {
  return __builtin_bit_cast(float, __builtin_amdgcn_ds_swizzle(__builtin_bit_cast(int, v), 0x401F));
}

// A-fragment (16x32 f16): src row-major [*, ld] halves, row m = lane&15.
// halves[0..7]=K(kb8+0..7), halves[8..15]=K(kb8+16..23), kb8 = (lane<16?0:8)
__device__ __forceinline__ AF load_afrag(const _Float16* row_kb8 /* = base + m*ld + k0 + kb8 */) {
  AF f;
  f.q[0] = *(const v4i*)(row_kb8);
  f.q[1] = *(const v4i*)(row_kb8 + 16);
  return f;
}
// B-fragment (32x16 f16): math B[k][n] taken from src[n][k] (K-contiguous),
// col n = lane&15, halves[0..7]=K(kb8+0..7), halves[8..15]=K(kb8+16..23), kb8 = (lane<16?0:8)
__device__ __forceinline__ AF load_bfrag(const _Float16* row_kb16 /* = base + n*ld + k0 + kb8 */) {
  AF f;
  f.q[0] = *(const v4i*)(row_kb16);
  f.q[1] = *(const v4i*)(row_kb16 + 16);
  return f;
}

// ---------------------------------------------------------------- convert f32 -> f16
__global__ __launch_bounds__(256) void f32_to_f16(const float* __restrict__ x,
                                                  _Float16* __restrict__ y, int n4) {
  int i = blockIdx.x * 256 + threadIdx.x;
  if (i >= n4) return;
  float4 v = ((const float4*)x)[i];
  int2 p;
  p.x = pkh(v.x, v.y);
  p.y = pkh(v.z, v.w);
  ((int2*)y)[i] = p;
}

// ---------------------------------------------------------------- W[1024x1024] f32 -> WT[n][k] f16
__global__ __launch_bounds__(256) void wtrans_f16(const float* __restrict__ W,
                                                  _Float16* __restrict__ WT) {
  __shared__ float sh[32][33];
  int tx = threadIdx.x & 31;
  int ty = threadIdx.x >> 5;  // 0..7
  int n0 = blockIdx.x * 32, k0 = blockIdx.y * 32;
#pragma unroll
  for (int r = 0; r < 4; r++)
    sh[ty * 4 + r][tx] = W[(size_t)(k0 + ty * 4 + r) * Dn + n0 + tx];
  __syncthreads();
#pragma unroll
  for (int r = 0; r < 4; r++)
    WT[(size_t)(n0 + ty * 4 + r) * Dn + k0 + tx] = (_Float16)sh[tx][ty * 4 + r];
}

// ---------------------------------------------------------------- C^T GEMM for q/k projections
// D[n][m] = sum_k WT[n][k]*X[m][k] = (X@W)^T ; out: [(b*H+h)*S + s]*64 + d, f16, *(acc+bias)*qs
__global__ __launch_bounds__(128) void gemm_ct_qk(const _Float16* __restrict__ X,
                                                  const _Float16* __restrict__ WT,
                                                  const float* __restrict__ bias,
                                                  _Float16* __restrict__ out, float qs) {
  const int lane = threadIdx.x & 31, wave = threadIdx.x >> 5;
  const int l15 = lane & 15;
  const bool hi = lane >= 16;
  const int nbase = blockIdx.x * 128 + (wave & 1) * 64;
  const int mbase = blockIdx.y * 128 + (wave >> 1) * 64;
  const int kb8 = hi ? 8 : 0, kb16 = hi ? 16 : 0;

  v8f acc[4][4] = {};
  for (int k0 = 0; k0 < Dn; k0 += 32) {
    AF a[4], bb[4];
#pragma unroll
    for (int i = 0; i < 4; i++)
      a[i] = load_afrag(WT + (size_t)(nbase + i * 16 + l15) * Dn + k0 + kb8);
#pragma unroll
    for (int j = 0; j < 4; j++)
      bb[j] = load_bfrag(X + (size_t)(mbase + j * 16 + l15) * Dn + k0 + kb8);
#pragma unroll
    for (int i = 0; i < 4; i++)
#pragma unroll
      for (int j = 0; j < 4; j++) acc[i][j] = wmma16(a[i], bb[j], acc[i][j]);
  }
  const int hh = nbase >> 6;  // whole wave covers exactly one head
#pragma unroll
  for (int i = 0; i < 4; i++) {
    const int r0 = i * 16 + (hi ? 8 : 0);
    float4 b0 = *(const float4*)(bias + nbase + r0);
    float4 b1 = *(const float4*)(bias + nbase + r0 + 4);
    float bb8[8] = {b0.x, b0.y, b0.z, b0.w, b1.x, b1.y, b1.z, b1.w};
#pragma unroll
    for (int j = 0; j < 4; j++) {
      int m = mbase + j * 16 + l15;
      int bi = m >> 11, s = m & (Sn - 1);
      _Float16* op = out + (((size_t)bi * Hn + hh) * Sn + s) * DH + r0;
      v4i pk;
#pragma unroll
      for (int e = 0; e < 4; e++)
        pk[e] = pkh((acc[i][j][2 * e] + bb8[2 * e]) * qs,
                    (acc[i][j][2 * e + 1] + bb8[2 * e + 1]) * qs);
      *(v4i*)op = pk;
    }
  }
}

// ---------------------------------------------------------------- C GEMM for v projection
// D[m][n] = sum_k X[m][k]*WT[n][k] ; out transposed per head: [(b*H+h)*64 + d]*S + s, f16
__global__ __launch_bounds__(128) void gemm_c_vT(const _Float16* __restrict__ X,
                                                 const _Float16* __restrict__ WT,
                                                 const float* __restrict__ bias,
                                                 _Float16* __restrict__ out) {
  const int lane = threadIdx.x & 31, wave = threadIdx.x >> 5;
  const int l15 = lane & 15;
  const bool hi = lane >= 16;
  const int nbase = blockIdx.x * 128 + (wave & 1) * 64;
  const int mbase = blockIdx.y * 128 + (wave >> 1) * 64;
  const int kb8 = hi ? 8 : 0, kb16 = hi ? 16 : 0;

  v8f acc[4][4] = {};
  for (int k0 = 0; k0 < Dn; k0 += 32) {
    AF a[4], bb[4];
#pragma unroll
    for (int i = 0; i < 4; i++)
      a[i] = load_afrag(X + (size_t)(mbase + i * 16 + l15) * Dn + k0 + kb8);
#pragma unroll
    for (int j = 0; j < 4; j++)
      bb[j] = load_bfrag(WT + (size_t)(nbase + j * 16 + l15) * Dn + k0 + kb8);
#pragma unroll
    for (int i = 0; i < 4; i++)
#pragma unroll
      for (int j = 0; j < 4; j++) acc[i][j] = wmma16(a[i], bb[j], acc[i][j]);
  }
#pragma unroll
  for (int j = 0; j < 4; j++) {
    int n = nbase + j * 16 + l15;
    int hh = n >> 6, d = n & 63;
    float bv = bias[n];
#pragma unroll
    for (int i = 0; i < 4; i++) {
      int r0 = i * 16 + (hi ? 8 : 0);
      int m0 = mbase + r0;
      int bi = m0 >> 11, s0 = m0 & (Sn - 1);
      _Float16* op = out + (((size_t)bi * Hn + hh) * DH + d) * Sn + s0;
      v4i pk;
#pragma unroll
      for (int e = 0; e < 4; e++)
        pk[e] = pkh(acc[i][j][2 * e] + bv, acc[i][j][2 * e + 1] + bv);
      *(v4i*)op = pk;
    }
  }
}

// ---------------------------------------------------------------- flash attention (per head)
// Computes S^T = K_h @ q_h^T so each lane owns one query column -> lane-local softmax.
__global__ __launch_bounds__(128) void attn_fused(const _Float16* __restrict__ qh,
                                                  const _Float16* __restrict__ kh,
                                                  const _Float16* __restrict__ vT,
                                                  const int* __restrict__ mask,
                                                  _Float16* __restrict__ attn_out) {
  const int lane = threadIdx.x & 31, wave = threadIdx.x >> 5;
  const int l15 = lane & 15;
  const bool hi = lane >= 16;
  const int b = blockIdx.z, h = blockIdx.y;
  const int q0 = blockIdx.x * 64 + wave * 16;

  const _Float16* qbh = qh + ((size_t)b * Hn + h) * Sn * DH;
  const _Float16* kbh = kh + ((size_t)b * Hn + h) * Sn * DH;
  const _Float16* vbh = vT + ((size_t)b * Hn + h) * DH * Sn;
  const int* mb = mask + (size_t)b * Sn;

  // q B-fragments (d 0..31 and 32..63), q pre-scaled by 1/8 in projection
  AF bq0, bq1;
  {
    const _Float16* p = qbh + (size_t)(q0 + l15) * DH + (hi ? 8 : 0);
    bq0.q[0] = *(const v4i*)p;        bq0.q[1] = *(const v4i*)(p + 16);
    bq1.q[0] = *(const v4i*)(p + 32); bq1.q[1] = *(const v4i*)(p + 48);
  }

  v8f o[4] = {};
  float m_run = -1e30f, l_run = 0.f;

  for (int kt = 0; kt < Sn; kt += 32) {
    // ---- scores^T chunk: 32 keys x 16 queries
    v8f ca = {}, cb = {};
    {
      const _Float16* kr = kbh + (size_t)(kt + l15) * DH + (hi ? 8 : 0);
      AF a;
      a.q[0] = *(const v4i*)kr;        a.q[1] = *(const v4i*)(kr + 16);
      ca = wmma16(a, bq0, ca);
      a.q[0] = *(const v4i*)(kr + 32); a.q[1] = *(const v4i*)(kr + 48);
      ca = wmma16(a, bq1, ca);
      kr += 16 * DH;
      a.q[0] = *(const v4i*)kr;        a.q[1] = *(const v4i*)(kr + 16);
      cb = wmma16(a, bq0, cb);
      a.q[0] = *(const v4i*)(kr + 32); a.q[1] = *(const v4i*)(kr + 48);
      cb = wmma16(a, bq1, cb);
    }
    if (kt + 32 < Sn) {
      __builtin_prefetch(kbh + (size_t)(kt + 32 + l15) * DH, 0, 0);
      __builtin_prefetch(vbh + (size_t)l15 * Sn + kt + 32, 0, 0);
    }
    // ---- mask -> s values (rows are keys; contiguous int loads)
    const int r0a = kt + (hi ? 8 : 0);
    v4i mk0 = *(const v4i*)(mb + r0a);
    v4i mk1 = *(const v4i*)(mb + r0a + 4);
    v4i mk2 = *(const v4i*)(mb + r0a + 16);
    v4i mk3 = *(const v4i*)(mb + r0a + 20);
    float sa[8], sb[8];
#pragma unroll
    for (int j = 0; j < 8; j++) {
      int ma = (j < 4) ? mk0[j & 3] : mk1[j & 3];
      int mc = (j < 4) ? mk2[j & 3] : mk3[j & 3];
      sa[j] = ma ? ca[j] : -1e30f;
      sb[j] = mc ? cb[j] : -1e30f;
    }
    // ---- online softmax (per query = per lane, partner lane holds other 16 keys)
    float cm = sa[0];
#pragma unroll
    for (int j = 0; j < 8; j++) { cm = fmaxf(cm, sa[j]); cm = fmaxf(cm, sb[j]); }
    cm = fmaxf(cm, swap16f(cm));
    float m_new = fmaxf(m_run, cm);
    float scale = __expf(m_run - m_new);
    float pa[8], pb_[8], psum = 0.f;
#pragma unroll
    for (int j = 0; j < 8; j++) {
      pa[j] = __expf(sa[j] - m_new);
      pb_[j] = __expf(sb[j] - m_new);
      psum += pa[j] + pb_[j];
    }
    psum += swap16f(psum);
    l_run = l_run * scale + psum;
    m_run = m_new;
#pragma unroll
    for (int t = 0; t < 4; t++)
#pragma unroll
      for (int j = 0; j < 8; j++) o[t][j] *= scale;

    // ---- build P^T B-fragment (keys x queries) via one SWAPX16 exchange
    int x[4], y[4];
#pragma unroll
    for (int j = 0; j < 4; j++) {
      x[j] = pkh(pa[2 * j], pa[2 * j + 1]);
      y[j] = pkh(pb_[2 * j], pb_[2 * j + 1]);
    }
    AF pbf;
#pragma unroll
    for (int j = 0; j < 4; j++) {
      int rcv = swap16i(hi ? x[j] : y[j]);
      pbf.q[0][j] = hi ? rcv : x[j];
      pbf.q[1][j] = hi ? y[j] : rcv;
    }
    // ---- O^T += V^T(chunk) @ P^T
#pragma unroll
    for (int t = 0; t < 4; t++) {
      const _Float16* vr = vbh + (size_t)(t * 16 + l15) * Sn + kt + (hi ? 16 : 0);
      AF av;
      av.q[0] = *(const v4i*)vr;
      av.q[1] = *(const v4i*)(vr + 8);
      o[t] = wmma16(av, pbf, o[t]);
    }
  }
  // ---- epilogue: divide by row-sum, pack, store as attn_out[b*S+q][h*64+d]
  const float inv = 1.0f / l_run;
#pragma unroll
  for (int t = 0; t < 4; t++) {
    v4i pk;
#pragma unroll
    for (int e = 0; e < 4; e++)
      pk[e] = pkh(o[t][2 * e] * inv, o[t][2 * e + 1] * inv);
    _Float16* op = attn_out + (size_t)(b * Sn + q0 + l15) * Dn + h * DH + t * 16 + (hi ? 8 : 0);
    *(v4i*)op = pk;
  }
}

// ---------------------------------------------------------------- final C^T GEMM, f32 out + bias
__global__ __launch_bounds__(128) void gemm_ct_out(const _Float16* __restrict__ X,
                                                   const _Float16* __restrict__ WT,
                                                   const float* __restrict__ bias,
                                                   float* __restrict__ out) {
  const int lane = threadIdx.x & 31, wave = threadIdx.x >> 5;
  const int l15 = lane & 15;
  const bool hi = lane >= 16;
  const int nbase = blockIdx.x * 128 + (wave & 1) * 64;
  const int mbase = blockIdx.y * 128 + (wave >> 1) * 64;
  const int kb8 = hi ? 8 : 0, kb16 = hi ? 16 : 0;

  v8f acc[4][4] = {};
  for (int k0 = 0; k0 < Dn; k0 += 32) {
    AF a[4], bb[4];
#pragma unroll
    for (int i = 0; i < 4; i++)
      a[i] = load_afrag(WT + (size_t)(nbase + i * 16 + l15) * Dn + k0 + kb8);
#pragma unroll
    for (int j = 0; j < 4; j++)
      bb[j] = load_bfrag(X + (size_t)(mbase + j * 16 + l15) * Dn + k0 + kb8);
#pragma unroll
    for (int i = 0; i < 4; i++)
#pragma unroll
      for (int j = 0; j < 4; j++) acc[i][j] = wmma16(a[i], bb[j], acc[i][j]);
  }
#pragma unroll
  for (int i = 0; i < 4; i++) {
    const int r0 = i * 16 + (hi ? 8 : 0);
    float4 b0 = *(const float4*)(bias + nbase + r0);
    float4 b1 = *(const float4*)(bias + nbase + r0 + 4);
#pragma unroll
    for (int j = 0; j < 4; j++) {
      int m = mbase + j * 16 + l15;
      float* op = out + (size_t)m * Dn + nbase + r0;
      float4 v0 = {acc[i][j][0] + b0.x, acc[i][j][1] + b0.y,
                   acc[i][j][2] + b0.z, acc[i][j][3] + b0.w};
      float4 v1 = {acc[i][j][4] + b1.x, acc[i][j][5] + b1.y,
                   acc[i][j][6] + b1.z, acc[i][j][7] + b1.w};
      *(float4*)op = v0;
      *((float4*)op + 1) = v1;
    }
  }
}

// ---------------------------------------------------------------- host launcher
extern "C" void kernel_launch(void* const* d_in, const int* in_sizes, int n_in,
                              void* d_out, int out_size, void* d_ws, size_t ws_size,
                              hipStream_t stream) {
  const float* Q  = (const float*)d_in[0];
  const float* K  = (const float*)d_in[1];
  const float* V  = (const float*)d_in[2];
  const int*   Mk = (const int*)d_in[3];
  const float* Wq = (const float*)d_in[4];
  const float* bq = (const float*)d_in[5];
  const float* Wk = (const float*)d_in[6];
  const float* bk = (const float*)d_in[7];
  const float* Wv = (const float*)d_in[8];
  const float* bv = (const float*)d_in[9];
  const float* Wo = (const float*)d_in[10];
  const float* bo = (const float*)d_in[11];
  float* out = (float*)d_out;

  const size_t ACT = (size_t)Bn * Sn * Dn * sizeof(_Float16);  // 8 MB
  const size_t WTS = (size_t)Dn * Dn * sizeof(_Float16);       // 2 MB
  char* p = (char*)d_ws;
  _Float16* Qh  = (_Float16*)p; p += ACT;
  _Float16* Kh  = (_Float16*)p; p += ACT;
  _Float16* Vh  = (_Float16*)p; p += ACT;
  _Float16* WqT = (_Float16*)p; p += WTS;
  _Float16* WkT = (_Float16*)p; p += WTS;
  _Float16* WvT = (_Float16*)p; p += WTS;
  _Float16* WoT = (_Float16*)p; p += WTS;
  _Float16* qhb = (_Float16*)p; p += ACT;   // [b][h][s][d], scaled by 1/8
  _Float16* khb = (_Float16*)p; p += ACT;   // [b][h][s][d]
  _Float16* vTb = (_Float16*)p; p += ACT;   // [b][h][d][s]
  _Float16* aoh = (_Float16*)p; p += ACT;   // attn out [b*s][dm]

  dim3 tB(256);
  dim3 tG(32, 32);
  wtrans_f16<<<tG, tB, 0, stream>>>(Wq, WqT);
  wtrans_f16<<<tG, tB, 0, stream>>>(Wk, WkT);
  wtrans_f16<<<tG, tB, 0, stream>>>(Wv, WvT);
  wtrans_f16<<<tG, tB, 0, stream>>>(Wo, WoT);

  const int n4 = Bn * Sn * Dn / 4;
  f32_to_f16<<<(n4 + 255) / 256, 256, 0, stream>>>(Q, Qh, n4);
  f32_to_f16<<<(n4 + 255) / 256, 256, 0, stream>>>(K, Kh, n4);
  f32_to_f16<<<(n4 + 255) / 256, 256, 0, stream>>>(V, Vh, n4);

  dim3 gG(Dn / 128, Bn * Sn / 128);  // (8, 32)
  gemm_ct_qk<<<gG, 128, 0, stream>>>(Qh, WqT, bq, qhb, 0.125f);  // 1/sqrt(64)
  gemm_ct_qk<<<gG, 128, 0, stream>>>(Kh, WkT, bk, khb, 1.0f);
  gemm_c_vT <<<gG, 128, 0, stream>>>(Vh, WvT, bv, vTb);

  attn_fused<<<dim3(Sn / 64, Hn, Bn), 128, 0, stream>>>(qhb, khb, vTb, Mk, aoh);

  gemm_ct_out<<<gG, 128, 0, stream>>>(aoh, WoT, bo, out);

  (void)in_sizes; (void)n_in; (void)out_size; (void)ws_size;
}
